// VanillaMPNN_51814485458973
// MI455X (gfx1250) — compile-verified, matrix operand order checked by reading, not run on hardware
//
#include <hip/hip_runtime.h>
#include <hip/hip_bf16.h>
#include <hip/hip_runtime_api.h>

// ---------------------------------------------------------------------------
// MPNN on MI455X (gfx1250).
//   * Big edge layer BN is computed analytically (a_j, beta_j) so the 4.3GB
//     Wedge tensor is NEVER materialized.
//   * Per-step NNConv is a fused bf16 WMMA GEMM against the L2-resident
//     U (128 x 16384, 4MB) with the ecc-contraction done through LDS.
// ---------------------------------------------------------------------------

typedef __attribute__((ext_vector_type(16))) __bf16 v16bf;
typedef __attribute__((ext_vector_type(8)))  float  v8f;

#define H      128
#define NNODE  2048
#define NEDGE  16384
#define NSTEP  6
#define EPSBN  1e-5f

// ---------------------------------------------------------------- utilities
__global__ void k_zero(float* __restrict__ p, int n) {
  int i = blockIdx.x * blockDim.x + threadIdx.x;
  if (i < n) p[i] = 0.f;
}

// Y[r,c] = X[r,:] . W[c,:] + b[c]
__global__ void k_linear(const float* __restrict__ X, const float* __restrict__ W,
                         const float* __restrict__ b, float* __restrict__ Y,
                         int R, int Cout, int K) {
  int idx = blockIdx.x * blockDim.x + threadIdx.x;
  if (idx >= R * Cout) return;
  int r = idx / Cout, c = idx % Cout;
  const float* x = X + (size_t)r * K;
  const float* w = W + (size_t)c * K;
  float acc = b[c];
  for (int k = 0; k < K; ++k) acc += x[k] * w[k];
  Y[idx] = acc;
}

// per-column mean and rsqrt(var+eps) over R rows (biased variance)
__global__ __launch_bounds__(256)
void k_colstats(const float* __restrict__ Y, int R, int Cc,
                float* __restrict__ mean, float* __restrict__ rstd) {
  int c = blockIdx.x;
  float s = 0.f, s2 = 0.f;
  for (int r = threadIdx.x; r < R; r += 256) {
    float v = Y[(size_t)r * Cc + c];
    s += v; s2 += v * v;
  }
  __shared__ float sh[512];
  sh[threadIdx.x] = s; sh[256 + threadIdx.x] = s2;
  __syncthreads();
  for (int o = 128; o > 0; o >>= 1) {
    if ((int)threadIdx.x < o) {
      sh[threadIdx.x] += sh[threadIdx.x + o];
      sh[256 + threadIdx.x] += sh[256 + threadIdx.x + o];
    }
    __syncthreads();
  }
  if (threadIdx.x == 0) {
    float mu = sh[0] / (float)R;
    float var = sh[256] / (float)R - mu * mu;
    mean[c] = mu;
    rstd[c] = rsqrtf(var + EPSBN);
  }
}

__global__ void k_bn(float* __restrict__ Y, const float* __restrict__ mean,
                     const float* __restrict__ rstd, const float* __restrict__ g,
                     const float* __restrict__ be, int n, int Cc, int relu) {
  int i = blockIdx.x * blockDim.x + threadIdx.x;
  if (i >= n) return;
  int c = i % Cc;
  float v = g[c] * (Y[i] - mean[c]) * rstd[c] + be[c];
  Y[i] = relu ? fmaxf(v, 0.f) : v;
}

__global__ void k_center(const float* __restrict__ Y, const float* __restrict__ mean,
                         float* __restrict__ O, int n, int Cc) {
  int i = blockIdx.x * blockDim.x + threadIdx.x;
  if (i >= n) return;
  O[i] = Y[i] - mean[i % Cc];
}

// C = ecc^T ecc / E  (128x128), 16x16 output tiles per block
__global__ __launch_bounds__(256)
void k_cov(const float* __restrict__ ecc, float* __restrict__ Cmat) {
  __shared__ float ea[32][16], eb[32][16];
  int a0 = (blockIdx.x >> 3) * 16, b0 = (blockIdx.x & 7) * 16;
  int al = threadIdx.x >> 4, bl = threadIdx.x & 15;
  float acc = 0.f;
  for (int r0 = 0; r0 < NEDGE; r0 += 32) {
    for (int t = threadIdx.x; t < 512; t += 256) {
      int rr = t >> 4, cc = t & 15;
      ea[rr][cc] = ecc[(size_t)(r0 + rr) * H + a0 + cc];
      eb[rr][cc] = ecc[(size_t)(r0 + rr) * H + b0 + cc];
    }
    __syncthreads();
#pragma unroll
    for (int rr = 0; rr < 32; ++rr) acc += ea[rr][al] * eb[rr][bl];
    __syncthreads();
  }
  Cmat[(a0 + al) * H + b0 + bl] = acc * (1.f / (float)NEDGE);
}

// a_j = g_j * rsqrt(w_j^T C w_j + eps), one block (128 thr) per j
__global__ __launch_bounds__(128)
void k_scale(const float* __restrict__ Cmat, const float* __restrict__ W4,
             const float* __restrict__ g4, float* __restrict__ ascale) {
  __shared__ float wsh[128], part[128];
  int j = blockIdx.x, p = threadIdx.x;
  wsh[p] = W4[(size_t)j * H + p];
  __syncthreads();
  const float* crow = Cmat + p * H;
  float t = 0.f;
  for (int q = 0; q < H; ++q) t += crow[q] * wsh[q];
  part[p] = t * wsh[p];
  __syncthreads();
  for (int o = 64; o > 0; o >>= 1) { if (p < o) part[p] += part[p + o]; __syncthreads(); }
  if (p == 0) ascale[j] = g4[j] * rsqrtf(part[0] + EPSBN);
}

// UT[col][i] = a[i*128+o] * W4[(i*128+o)*128 + k],  col = k*128+o   (bf16, 4MB)
__global__ void k_build_UT(const float* __restrict__ W4, const float* __restrict__ ascale,
                           __bf16* __restrict__ UT) {
  int idx = blockIdx.x * blockDim.x + threadIdx.x;   // 16384*128
  if (idx >= NEDGE * H) return;
  int col = idx / H, i = idx % H;
  int k = col >> 7, o = col & 127;
  int j = i * H + o;
  UT[(size_t)col * H + i] = (__bf16)(ascale[j] * W4[(size_t)j * H + k]);
}

__global__ void k_count(const int* __restrict__ dst, float* __restrict__ deg, int E) {
  int e = blockIdx.x * blockDim.x + threadIdx.x;
  if (e < E) atomicAdd(&deg[dst[e]], 1.f);
}

// nodebeta[n,o] = sum_i h[n,i] * beta4[i*128+o]
__global__ void k_nodebeta(const float* __restrict__ hst, const float* __restrict__ be4,
                           float* __restrict__ nb) {
  int idx = blockIdx.x * blockDim.x + threadIdx.x;   // 2048*128
  int n = idx >> 7, o = idx & 127;
  const float* hr = hst + (size_t)n * H;
  float acc = 0.f;
  for (int i = 0; i < H; ++i) acc += hr[i] * be4[(size_t)i * H + o];
  nb[idx] = acc;
}

// -------------------------------------------------------------- fused NNConv
// Per block: TE=16 edges.  P(16x2048 f32) = Dbf(16x128) @ U_chunk via WMMA,
// then msg[e,o] += sum_kl ecc[e, kc*16+kl] * P[e, kl*128+o].  8 chunks.
// Epilogue: += nodebeta[src], atomic scatter to agg[dst].
#define NNCONV_LDS (131072 + 4096 + 8192 + 128)

__global__ __launch_bounds__(256)
void k_nnconv(const float* __restrict__ node, const float* __restrict__ ecc,
              const __bf16* __restrict__ UT, const float* __restrict__ nodebeta,
              const int* __restrict__ src, const int* __restrict__ dst,
              float* __restrict__ agg) {
  extern __shared__ char smem[];
  float*  P   = (float*)smem;                       // [16][2048]  128KB
  __bf16* Dl  = (__bf16*)(smem + 131072);           // [16][128]   4KB
  float*  El  = (float*)(smem + 135168);            // [16][128]   8KB
  int*   sSrc = (int*)(smem + 143360);
  int*   sDst = (int*)(smem + 143424);

  const int tid = threadIdx.x;
  const int e0 = blockIdx.x * 16;
  if (tid < 16) { sSrc[tid] = src[e0 + tid]; sDst[tid] = dst[e0 + tid]; }
  __syncthreads();

  { // gather src node rows (f32 -> bf16) and ecc rows into LDS
    int e = tid >> 4, c0 = (tid & 15) * 8;
    const float* nr = node + (size_t)sSrc[e] * H + c0;
    const float* er = ecc + (size_t)(e0 + e) * H + c0;
#pragma unroll
    for (int j = 0; j < 8; ++j) {
      Dl[e * H + c0 + j] = (__bf16)nr[j];
      El[e * H + c0 + j] = er[j];
    }
  }
  __syncthreads();

  const int lane = tid & 31;
  const int wv   = tid >> 5;        // 8 waves
  const int l15  = lane & 15;
  const int hi   = lane >> 4;

  // A fragments: 16x128 (M x K), K split into 4 slices of 32 (ISA 7.12.2 layout)
  v16bf afrag[4];
  {
    const __bf16* drow = Dl + l15 * H;
#pragma unroll
    for (int ks = 0; ks < 4; ++ks) {
      union { v16bf v; uint4 q[2]; } u;
      u.q[0] = *(const uint4*)(drow + ks * 32 + hi * 8);
      u.q[1] = *(const uint4*)(drow + ks * 32 + 16 + hi * 8);
      afrag[ks] = u.v;
    }
  }

  float acc8[8];
#pragma unroll
  for (int j = 0; j < 8; ++j) acc8[j] = 0.f;

  for (int kc = 0; kc < 8; ++kc) {
    const __bf16* Ub = UT + (size_t)(kc * 2048) * H;
    if (kc < 7) __builtin_prefetch(Ub + (size_t)2048 * H, 0, 1);  // next chunk hint

#pragma unroll 1
    for (int nt = 0; nt < 16; ++nt) {            // 16 N-tiles per wave
      const int colBase = (wv * 16 + nt) * 16;
      v8f c = {};
#pragma unroll
      for (int ks = 0; ks < 4; ++ks) {
        const __bf16* bp = Ub + (size_t)(colBase + l15) * H + ks * 32 + hi * 8;
        union { v16bf v; uint4 q[2]; } ub;
        ub.q[0] = *(const uint4*)(bp);
        ub.q[1] = *(const uint4*)(bp + 16);
        c = __builtin_amdgcn_wmma_f32_16x16x32_bf16(false, afrag[ks], false, ub.v,
                                                    (short)0, c, false, false);
      }
      // C layout: VGPR r -> row (r + 8*hi), column = l15
#pragma unroll
      for (int r = 0; r < 8; ++r) P[(r + 8 * hi) * 2048 + colBase + l15] = c[r];
    }
    __syncthreads();

    { // msg accumulation: thread -> (edge e, 8 outputs at o0)
      int e = tid >> 4, o0 = (tid & 15) * 8;
#pragma unroll
      for (int kl = 0; kl < 16; ++kl) {
        float s = El[e * H + kc * 16 + kl];
        const float* pp = P + e * 2048 + kl * H + o0;
#pragma unroll
        for (int j = 0; j < 8; ++j) acc8[j] += s * pp[j];
      }
    }
    __syncthreads();
  }

  { // += x.beta term, scatter-add to agg[dst]
    int e = tid >> 4, o0 = (tid & 15) * 8;
    const float* nb = nodebeta + (size_t)sSrc[e] * H + o0;
    float* ag = agg + (size_t)sDst[e] * H + o0;
#pragma unroll
    for (int j = 0; j < 8; ++j) atomicAdd(&ag[j], acc8[j] + nb[j]);
  }
}

// m = relu(agg / max(deg,1) + conv_bias), in-place
__global__ void k_postagg(float* __restrict__ agg, const float* __restrict__ deg,
                          const float* __restrict__ cbias) {
  int idx = blockIdx.x * blockDim.x + threadIdx.x;  // 2048*128
  int n = idx >> 7, o = idx & 127;
  float d = fmaxf(deg[n], 1.f);
  agg[idx] = fmaxf(agg[idx] / d + cbias[o], 0.f);
}

// GRU cell, 2 nodes per block (PyTorch gate order r,z,n)
__global__ __launch_bounds__(256)
void k_gru(const float* __restrict__ m, float* __restrict__ h,
           const float* __restrict__ Wih, const float* __restrict__ Whh,
           const float* __restrict__ bih, const float* __restrict__ bhh) {
  __shared__ float mm[2][128], hh[2][128];
  int ln = threadIdx.x >> 7, u = threadIdx.x & 127;
  int n = blockIdx.x * 2 + ln;
  mm[ln][u] = m[(size_t)n * H + u];
  hh[ln][u] = h[(size_t)n * H + u];
  __syncthreads();
  float gi[3], gh[3];
#pragma unroll
  for (int g3 = 0; g3 < 3; ++g3) {
    const float* wi = Wih + (size_t)(g3 * H + u) * H;
    const float* wh = Whh + (size_t)(g3 * H + u) * H;
    float ai = bih[g3 * H + u], ah = bhh[g3 * H + u];
    for (int k = 0; k < H; ++k) { ai += wi[k] * mm[ln][k]; ah += wh[k] * hh[ln][k]; }
    gi[g3] = ai; gh[g3] = ah;
  }
  float r  = 1.f / (1.f + __expf(-(gi[0] + gh[0])));
  float z  = 1.f / (1.f + __expf(-(gi[1] + gh[1])));
  float nn = tanhf(gi[2] + r * gh[2]);
  h[(size_t)n * H + u] = (1.f - z) * nn + z * hh[ln][u];
}

__global__ void k_out(const float* __restrict__ y1, const float* __restrict__ Wo,
                      const float* __restrict__ bo, float* __restrict__ out, int Nn) {
  int n = blockIdx.x * blockDim.x + threadIdx.x;
  if (n >= Nn) return;
  float acc = bo[0];
  for (int k = 0; k < H; ++k) acc += y1[(size_t)n * H + k] * Wo[k];
  out[n] = acc;
}

// ---------------------------------------------------------------------------
extern "C" void kernel_launch(void* const* d_in, const int* in_sizes, int n_in,
                              void* d_out, int out_size, void* d_ws, size_t ws_size,
                              hipStream_t stream) {
  const float* node_feats = (const float*)d_in[0];
  const float* edge_feats = (const float*)d_in[1];
  const int*   src        = (const int*)d_in[2];
  const int*   dst        = (const int*)d_in[3];
  // pre[0], pre[1]
  const float *Wq0 = (const float*)d_in[4],  *bq0 = (const float*)d_in[5],
              *gq0 = (const float*)d_in[6],  *eq0 = (const float*)d_in[7];
  const float *Wq1 = (const float*)d_in[8],  *bq1 = (const float*)d_in[9],
              *gq1 = (const float*)d_in[10], *eq1 = (const float*)d_in[11];
  // edge[0..3]
  const float *We0 = (const float*)d_in[12], *be0 = (const float*)d_in[13],
              *ge0 = (const float*)d_in[14], *ee0 = (const float*)d_in[15];
  const float *We1 = (const float*)d_in[16], *be1 = (const float*)d_in[17],
              *ge1 = (const float*)d_in[18], *ee1 = (const float*)d_in[19];
  const float *We2 = (const float*)d_in[20], *be2 = (const float*)d_in[21],
              *ge2 = (const float*)d_in[22], *ee2 = (const float*)d_in[23];
  const float *W4  = (const float*)d_in[24];                 // b4 (d_in[25]) cancels in BN
  const float *g4  = (const float*)d_in[26], *beta4 = (const float*)d_in[27];
  const float *cbias = (const float*)d_in[28];
  const float *Wih = (const float*)d_in[29], *Whh = (const float*)d_in[30],
              *bih = (const float*)d_in[31], *bhh = (const float*)d_in[32];
  const float *Wp  = (const float*)d_in[33], *bp = (const float*)d_in[34],
              *gp  = (const float*)d_in[35], *bep = (const float*)d_in[36],
              *Wo  = (const float*)d_in[37], *bo = (const float*)d_in[38];
  (void)n_in; (void)in_sizes; (void)out_size; (void)ws_size;

  char* ws = (char*)d_ws;
  size_t off = 0;
  auto alloc = [&](size_t bytes) -> void* {
    void* p = ws + off;
    off = (off + bytes + 255) & ~(size_t)255;
    return p;
  };
  float* hstate   = (float*)alloc((size_t)NNODE * H * 4);
  float* bufm     = (float*)alloc((size_t)NNODE * H * 4);
  float* tmpA     = (float*)alloc((size_t)NNODE * 64 * 4);
  float* eb0      = (float*)alloc((size_t)NEDGE * H * 4);
  float* eb1      = (float*)alloc((size_t)NEDGE * H * 4);   // ecc lives here
  __bf16* UT      = (__bf16*)alloc((size_t)NEDGE * H * 2);
  float* ascale   = (float*)alloc((size_t)NEDGE * 4);
  float* Cmat     = (float*)alloc((size_t)H * H * 4);
  float* meanv    = (float*)alloc(H * 4);
  float* rstdv    = (float*)alloc(H * 4);
  float* nodebeta = (float*)alloc((size_t)NNODE * H * 4);
  float* deg      = (float*)alloc(NNODE * 4);
  float* y1       = (float*)alloc((size_t)NNODE * H * 4);

  auto lbn = [&](const float* X, const float* W, const float* b, const float* g,
                 const float* be, float* Y, int R, int Cout, int K, int relu) {
    int n = R * Cout;
    k_linear<<<(n + 255) / 256, 256, 0, stream>>>(X, W, b, Y, R, Cout, K);
    k_colstats<<<Cout, 256, 0, stream>>>(Y, R, Cout, meanv, rstdv);
    k_bn<<<(n + 255) / 256, 256, 0, stream>>>(Y, meanv, rstdv, g, be, n, Cout, relu);
  };

  // ---- preprocess MLP -> hstate (node==hidden)
  lbn(node_feats, Wq0, bq0, gq0, eq0, tmpA, NNODE, 64, 32, 1);
  lbn(tmpA, Wq1, bq1, gq1, eq1, hstate, NNODE, 128, 64, 1);

  // ---- edge MLP layers 0..2 (+relu) -> eb0
  lbn(edge_feats, We0, be0, ge0, ee0, eb0, NEDGE, 32, 16, 1);
  lbn(eb0, We1, be1, ge1, ee1, eb1, NEDGE, 64, 32, 1);
  lbn(eb1, We2, be2, ge2, ee2, eb0, NEDGE, 128, 64, 1);

  // ---- analytic BN for layer 3: ecc, C, a_j, U
  k_colstats<<<H, 256, 0, stream>>>(eb0, NEDGE, H, meanv, rstdv);
  k_center<<<(NEDGE * H + 255) / 256, 256, 0, stream>>>(eb0, meanv, eb1, NEDGE * H, H);
  k_cov<<<64, 256, 0, stream>>>(eb1, Cmat);
  k_scale<<<NEDGE, 128, 0, stream>>>(Cmat, W4, g4, ascale);
  k_build_UT<<<(NEDGE * H + 255) / 256, 256, 0, stream>>>(W4, ascale, UT);

  // ---- degree
  k_zero<<<(NNODE + 255) / 256, 256, 0, stream>>>(deg, NNODE);
  k_count<<<(NEDGE + 255) / 256, 256, 0, stream>>>(dst, deg, NEDGE);

  // ---- 6 message-passing steps
  for (int t = 0; t < NSTEP; ++t) {
    k_zero<<<(NNODE * H + 255) / 256, 256, 0, stream>>>(bufm, NNODE * H);
    k_nodebeta<<<(NNODE * H) / 256, 256, 0, stream>>>(hstate, beta4, nodebeta);
    k_nnconv<<<NEDGE / 16, 256, NNCONV_LDS, stream>>>(hstate, eb1, UT, nodebeta,
                                                      src, dst, bufm);
    k_postagg<<<(NNODE * H) / 256, 256, 0, stream>>>(bufm, deg, cbias);
    k_gru<<<NNODE / 2, 256, 0, stream>>>(bufm, hstate, Wih, Whh, bih, bhh);
  }

  // ---- readout
  lbn(hstate, Wp, bp, gp, bep, y1, NNODE, 128, 128, 1);
  k_out<<<(NNODE + 255) / 256, 256, 0, stream>>>(y1, Wo, bo, (float*)d_out, NNODE);
}
